// AttentionPairBias_14456859918336
// MI455X (gfx1250) — matrix operand order repaired and checked by reading, not compiled, not run on hardware
//
#include <hip/hip_runtime.h>

constexpr int kN  = 1024;   // sequence length
constexpr int kCQ = 768;    // token channels
constexpr int kCZ = 128;    // pair channels
constexpr int kH  = 16;     // heads
constexpr int kD  = 48;     // head dim
constexpr int kHD = 768;    // H*D
constexpr float kEPS = 1e-5f;
constexpr float kINF = 1.0e9f;

typedef __attribute__((ext_vector_type(16))) __bf16 bf16x16;
typedef __attribute__((ext_vector_type(8)))  float  floatx8;

union Frag16 {
  bf16x16 v;
  unsigned int x[8];
  unsigned short u[16];
};

__device__ __forceinline__ unsigned short f2bf(float f) {
  unsigned int u = __float_as_uint(f);
  u += 0x7FFFu + ((u >> 16) & 1u);   // round-to-nearest-even
  return (unsigned short)(u >> 16);
}

__device__ __forceinline__ unsigned int ldpair(const unsigned short* p) {
  return *reinterpret_cast<const unsigned int*>(p);
}

__device__ __forceinline__ float wred_sum(float v) {
#pragma unroll
  for (int m = 16; m > 0; m >>= 1) v += __shfl_xor(v, m, 32);
  return v;
}
__device__ __forceinline__ float wred_max(float v) {
#pragma unroll
  for (int m = 16; m > 0; m >>= 1) v = fmaxf(v, __shfl_xor(v, m, 32));
  return v;
}

__device__ __forceinline__ floatx8 wmma_bf16(const Frag16& A, const Frag16& B, floatx8 C) {
  return __builtin_amdgcn_wmma_f32_16x16x32_bf16(
      /*neg_a=*/false, A.v, /*neg_b=*/false, B.v,
      /*c_mod=*/(short)0, C, /*reuse_a=*/false, /*reuse_b=*/false);
}

// ---------------------------------------------------------------------------
// Weight transpose + fp32 -> bf16.  w is [K][Nn] row-major, wt is [Nn][K].
__global__ __launch_bounds__(256) void k_wconv(const float* __restrict__ w,
                                               unsigned short* __restrict__ wt,
                                               int K, int Nn) {
  int t = blockIdx.x * blockDim.x + threadIdx.x;
  if (t >= K * Nn) return;
  int n = t / K;
  int k = t - n * K;
  wt[(size_t)n * K + k] = f2bf(w[(size_t)k * Nn + n]);
}

// ---------------------------------------------------------------------------
// LayerNorm over token embedding rows -> bf16.
__global__ __launch_bounds__(256) void k_ln_a(const float* __restrict__ a,
                                              const float* __restrict__ g,
                                              const float* __restrict__ b,
                                              unsigned short* __restrict__ an) {
  __shared__ float s1[256], s2[256];
  const int row = blockIdx.x, tid = threadIdx.x;
  const float* ar = a + (size_t)row * kCQ;
  float x0 = ar[tid], x1 = ar[tid + 256], x2 = ar[tid + 512];
  s1[tid] = x0 + x1 + x2;
  s2[tid] = x0 * x0 + x1 * x1 + x2 * x2;
  __syncthreads();
  for (int s = 128; s > 0; s >>= 1) {
    if (tid < s) { s1[tid] += s1[tid + s]; s2[tid] += s2[tid + s]; }
    __syncthreads();
  }
  float mu  = s1[0] * (1.0f / kCQ);
  float var = s2[0] * (1.0f / kCQ) - mu * mu;
  float rs  = rsqrtf(var + kEPS);
  unsigned short* orow = an + (size_t)row * kCQ;
  orow[tid]       = f2bf((x0 - mu) * rs * g[tid]       + b[tid]);
  orow[tid + 256] = f2bf((x1 - mu) * rs * g[tid + 256] + b[tid + 256]);
  orow[tid + 512] = f2bf((x2 - mu) * rs * g[tid + 512] + b[tid + 512]);
}

// ---------------------------------------------------------------------------
// Fused Q/K/V/G projection: an[1024,768] (bf16) x W^T -> 16x16 tile per wave.
__global__ __launch_bounds__(32) void k_proj(const unsigned short* __restrict__ an,
                                             const unsigned short* __restrict__ wqt,
                                             const unsigned short* __restrict__ wkt,
                                             const unsigned short* __restrict__ wvt,
                                             const unsigned short* __restrict__ wgt,
                                             const float* __restrict__ b_g,
                                             unsigned short* __restrict__ qb,
                                             unsigned short* __restrict__ kb,
                                             unsigned short* __restrict__ vt,
                                             float* __restrict__ gate) {
  const int lane = threadIdx.x;
  const int half = lane >> 4;
  const int nn   = lane & 15;
  const int n0   = blockIdx.x * 16;
  const int m0   = blockIdx.y * 16;
  const int mat  = blockIdx.z;
  const unsigned short* wt = (mat == 0) ? wqt : (mat == 1) ? wkt : (mat == 2) ? wvt : wgt;

  floatx8 acc = {0.f, 0.f, 0.f, 0.f, 0.f, 0.f, 0.f, 0.f};
#pragma unroll 1
  for (int kk = 0; kk < kCQ; kk += 32) {
    Frag16 A, B;
#pragma unroll
    for (int v = 0; v < 8; ++v) {
      int ka = (v < 4) ? (2 * v + 8 * half) : (16 + 2 * (v - 4) + 8 * half);
      A.x[v] = ldpair(an + (size_t)(m0 + nn) * kCQ + kk + ka);
      int kbk = (v < 4) ? (2 * v + 8 * half) : (16 + 2 * (v - 4) + 8 * half);
      B.x[v] = ldpair(wt + (size_t)(n0 + nn) * kCQ + kk + kbk);
    }
    acc = wmma_bf16(A, B, acc);
  }
  const float qscale = 0.14433756729740643f;  // 1/sqrt(48)
  const int col = n0 + nn;
  if (mat == 0) {
#pragma unroll
    for (int cf = 0; cf < 8; ++cf) {
      int row = m0 + cf + 8 * half;
      qb[(size_t)row * kHD + col] = f2bf(acc[cf] * qscale);
    }
  } else if (mat == 1) {
#pragma unroll
    for (int cf = 0; cf < 8; ++cf) {
      int row = m0 + cf + 8 * half;
      kb[(size_t)row * kHD + col] = f2bf(acc[cf]);
    }
  } else if (mat == 2) {
#pragma unroll
    for (int cf = 0; cf < 8; ++cf) {
      int row = m0 + cf + 8 * half;
      vt[(size_t)col * kN + row] = f2bf(acc[cf]);      // v transposed: [HD][N]
    }
  } else {
    const float bg = b_g[col];
#pragma unroll
    for (int cf = 0; cf < 8; ++cf) {
      int row = m0 + cf + 8 * half;
      gate[(size_t)row * kHD + col] = 1.0f / (1.0f + __expf(-(acc[cf] + bg)));
    }
  }
}

// ---------------------------------------------------------------------------
// Pair bias: LayerNorm(z[i,j,:]) . w_z -> pb[h][i][j].  One wave per vector.
__global__ __launch_bounds__(256) void k_pairbias(const float* __restrict__ z,
                                                  const float* __restrict__ g_z,
                                                  const float* __restrict__ b_z,
                                                  const float* __restrict__ w_z,
                                                  float* __restrict__ pb) {
  const int lane = threadIdx.x & 31;
  const int wave = threadIdx.x >> 5;
  const size_t vec = (size_t)blockIdx.x * 8 + wave;
  const int i = (int)(vec >> 10);
  const int j = (int)(vec & 1023);
  const float4 t = reinterpret_cast<const float4*>(z + vec * kCZ)[lane];
  float s = t.x + t.y + t.z + t.w;
  float q = t.x * t.x + t.y * t.y + t.z * t.z + t.w * t.w;
  s = wred_sum(s);
  q = wred_sum(q);
  const float mu  = s * (1.0f / kCZ);
  const float var = q * (1.0f / kCZ) - mu * mu;
  const float rs  = rsqrtf(var + kEPS);
  float acc[kH];
#pragma unroll
  for (int h = 0; h < kH; ++h) acc[h] = 0.f;
  const float zv[4] = {t.x, t.y, t.z, t.w};
#pragma unroll
  for (int e = 0; e < 4; ++e) {
    int c = 4 * lane + e;
    float nv = (zv[e] - mu) * rs * g_z[c] + b_z[c];
#pragma unroll
    for (int h = 0; h < kH; ++h) acc[h] += nv * w_z[c * kH + h];
  }
#pragma unroll
  for (int h = 0; h < kH; ++h) {
    float tot = wred_sum(acc[h]);
    if (lane == h) pb[((size_t)h << 20) + ((size_t)i << 10) + j] = tot;
  }
}

// ---------------------------------------------------------------------------
// Attention: 8 waves cooperate on one (head, 16-query block).
// Waves split k-tiles (scores), rows (softmax), K-chunks (PV); partial O
// fragments combined with LDS float atomics.
__global__ __launch_bounds__(256) void k_attn(const unsigned short* __restrict__ qb,
                                              const unsigned short* __restrict__ kbuf,
                                              const unsigned short* __restrict__ vt,
                                              const float* __restrict__ pb,
                                              const float* __restrict__ mask,
                                              const float* __restrict__ gate,
                                              unsigned short* __restrict__ og) {
  __shared__ float sc[16 * kN];       // 64 KB score/probability tile
  __shared__ float obuf[16 * kD];     // 3 KB partial-O accumulator
  const int tid  = threadIdx.x;
  const int lane = tid & 31;
  const int wave = tid >> 5;
  const int half = lane >> 4;
  const int nn   = lane & 15;
  const int h    = blockIdx.y;
  const int q0   = blockIdx.x * 16;

  // zero the partial-O accumulator (16*48 = 768 floats)
  obuf[tid] = 0.f;
  obuf[tid + 256] = 0.f;
  obuf[tid + 512] = 0.f;

  // Q fragments (K = 48, padded to 64 with zeros) — identical in every wave.
  Frag16 qa0, qa1;
  const unsigned short* qrow = qb + (size_t)(q0 + nn) * kHD + h * kD;
#pragma unroll
  for (int v = 0; v < 8; ++v) {
    int ka = (v < 4) ? (2 * v + 8 * half) : (16 + 2 * (v - 4) + 8 * half);
    qa0.x[v] = ldpair(qrow + ka);
    qa1.x[v] = (v < 4) ? ldpair(qrow + 32 + ka) : 0u;   // d in [32,48) real, rest zero
  }
  float mrow[8];
#pragma unroll
  for (int cf = 0; cf < 8; ++cf) mrow[cf] = mask[q0 + cf + 8 * half];

  // ---- scores = q.kT + mask bias + pair bias (waves split the 64 k-tiles) ----
  for (int jb = wave; jb < kN / 16; jb += 8) {
    const int j0 = jb * 16;
    Frag16 b0, b1;
    const unsigned short* krow = kbuf + (size_t)(j0 + nn) * kHD + h * kD;
#pragma unroll
    for (int v = 0; v < 8; ++v) {
      int kk = (v < 4) ? (2 * v + 8 * half) : (16 + 2 * (v - 4) + 8 * half);
      b0.x[v] = ldpair(krow + kk);
      b1.x[v] = (v < 4) ? ldpair(krow + 32 + kk) : 0u;  // d 32..47 / zero pad
    }
    floatx8 acc = {0.f, 0.f, 0.f, 0.f, 0.f, 0.f, 0.f, 0.f};
    acc = wmma_bf16(qa0, b0, acc);
    acc = wmma_bf16(qa1, b1, acc);
    float mj = mask[j0 + nn];
#pragma unroll
    for (int cf = 0; cf < 8; ++cf) {
      int M = cf + 8 * half;
      float bias = kINF * (mrow[cf] * mj - 1.0f);
      float pbv  = pb[((size_t)h << 20) + ((size_t)(q0 + M) << 10) + (j0 + nn)];
      sc[M * kN + j0 + nn] = acc[cf] + bias + pbv;
    }
  }
  __syncthreads();

  // ---- softmax: 16 rows, 2 rows per wave ----
  for (int r = wave * 2; r < wave * 2 + 2; ++r) {
    float mx = -kINF;
    for (int c = lane; c < kN; c += 32) mx = fmaxf(mx, sc[r * kN + c]);
    mx = wred_max(mx);
    float sum = 0.f;
    for (int c = lane; c < kN; c += 32) {
      float e = __expf(sc[r * kN + c] - mx);
      sc[r * kN + c] = e;
      sum += e;
    }
    sum = wred_sum(sum);
    float inv = 1.0f / sum;
    for (int c = lane; c < kN; c += 32) sc[r * kN + c] *= inv;
  }
  __syncthreads();

  // ---- o = p @ v (waves split the 32 K-chunks; N-dim 48 = 3 groups of 16) ----
  floatx8 o0 = {0.f, 0.f, 0.f, 0.f, 0.f, 0.f, 0.f, 0.f};
  floatx8 o1 = o0, o2 = o0;
  for (int jb = wave; jb < kN / 32; jb += 8) {
    Frag16 pa;
#pragma unroll
    for (int v = 0; v < 8; ++v) {
      int ka = (v < 4) ? (2 * v + 8 * half) : (16 + 2 * (v - 4) + 8 * half);
      int j = jb * 32 + ka;
      unsigned int lo = f2bf(sc[nn * kN + j]);
      unsigned int hi = f2bf(sc[nn * kN + j + 1]);
      pa.x[v] = lo | (hi << 16);
    }
    Frag16 vb0, vb1, vb2;
#pragma unroll
    for (int v = 0; v < 8; ++v) {
      int kk = (v < 4) ? (2 * v + 8 * half) : (16 + 2 * (v - 4) + 8 * half);
      int j = jb * 32 + kk;
      vb0.x[v] = ldpair(vt + (size_t)(h * kD + nn) * kN + j);
      vb1.x[v] = ldpair(vt + (size_t)(h * kD + 16 + nn) * kN + j);
      vb2.x[v] = ldpair(vt + (size_t)(h * kD + 32 + nn) * kN + j);
    }
    o0 = wmma_bf16(pa, vb0, o0);
    o1 = wmma_bf16(pa, vb1, o1);
    o2 = wmma_bf16(pa, vb2, o2);
  }

  // ---- combine partial O fragments across waves (ds_add_f32) ----
#pragma unroll
  for (int cf = 0; cf < 8; ++cf) {
    int M = cf + 8 * half;
    atomicAdd(&obuf[M * kD + nn],      o0[cf]);
    atomicAdd(&obuf[M * kD + nn + 16], o1[cf]);
    atomicAdd(&obuf[M * kD + nn + 32], o2[cf]);
  }
  __syncthreads();

  // ---- gate + store bf16 ----
#pragma unroll
  for (int e = tid; e < 16 * kD; e += 256) {
    int M = e / kD, c = e - M * kD;
    int row = q0 + M, col = h * kD + c;
    og[(size_t)row * kHD + col] = f2bf(obuf[e] * gate[(size_t)row * kHD + col]);
  }
}

// ---------------------------------------------------------------------------
// Output projection: og[1024,768] @ w_o + b_o -> fp32 out.
__global__ __launch_bounds__(32) void k_outgemm(const unsigned short* __restrict__ og,
                                                const unsigned short* __restrict__ wot,
                                                const float* __restrict__ b_o,
                                                float* __restrict__ out) {
  const int lane = threadIdx.x;
  const int half = lane >> 4;
  const int nn   = lane & 15;
  const int n0   = blockIdx.x * 16;
  const int m0   = blockIdx.y * 16;

  floatx8 acc = {0.f, 0.f, 0.f, 0.f, 0.f, 0.f, 0.f, 0.f};
#pragma unroll 1
  for (int kk = 0; kk < kHD; kk += 32) {
    Frag16 A, B;
#pragma unroll
    for (int v = 0; v < 8; ++v) {
      int ka = (v < 4) ? (2 * v + 8 * half) : (16 + 2 * (v - 4) + 8 * half);
      A.x[v] = ldpair(og + (size_t)(m0 + nn) * kHD + kk + ka);
      int kbk = (v < 4) ? (2 * v + 8 * half) : (16 + 2 * (v - 4) + 8 * half);
      B.x[v] = ldpair(wot + (size_t)(n0 + nn) * kHD + kk + kbk);
    }
    acc = wmma_bf16(A, B, acc);
  }
  const float bo = b_o[n0 + nn];
#pragma unroll
  for (int cf = 0; cf < 8; ++cf) {
    int row = m0 + cf + 8 * half;
    out[(size_t)row * kCQ + n0 + nn] = acc[cf] + bo;
  }
}

// ---------------------------------------------------------------------------
extern "C" void kernel_launch(void* const* d_in, const int* in_sizes, int n_in,
                              void* d_out, int out_size, void* d_ws, size_t ws_size,
                              hipStream_t stream) {
  const float* a    = (const float*)d_in[0];
  const float* z    = (const float*)d_in[1];
  const float* mask = (const float*)d_in[2];
  const float* g_a  = (const float*)d_in[3];
  const float* b_a  = (const float*)d_in[4];
  const float* g_z  = (const float*)d_in[5];
  const float* b_z  = (const float*)d_in[6];
  const float* w_z  = (const float*)d_in[7];
  const float* w_q  = (const float*)d_in[8];
  const float* w_k  = (const float*)d_in[9];
  const float* w_v  = (const float*)d_in[10];
  const float* w_g  = (const float*)d_in[11];
  const float* b_g  = (const float*)d_in[12];
  const float* w_o  = (const float*)d_in[13];
  const float* b_o  = (const float*)d_in[14];
  float* out = (float*)d_out;
  (void)in_sizes; (void)n_in; (void)out_size; (void)ws_size;

  char* base = (char*)d_ws;
  size_t off = 0;
  auto alloc = [&](size_t bytes) -> char* {
    char* p = base + off;
    off += (bytes + 255) & ~(size_t)255;
    return p;
  };
  unsigned short* wqt  = (unsigned short*)alloc((size_t)kCQ * kHD * 2);
  unsigned short* wkt  = (unsigned short*)alloc((size_t)kCQ * kHD * 2);
  unsigned short* wvt  = (unsigned short*)alloc((size_t)kCQ * kHD * 2);
  unsigned short* wgt  = (unsigned short*)alloc((size_t)kCQ * kHD * 2);
  unsigned short* wot  = (unsigned short*)alloc((size_t)kHD * kCQ * 2);
  unsigned short* an   = (unsigned short*)alloc((size_t)kN * kCQ * 2);
  unsigned short* qb   = (unsigned short*)alloc((size_t)kN * kHD * 2);
  unsigned short* kb   = (unsigned short*)alloc((size_t)kN * kHD * 2);
  unsigned short* vt   = (unsigned short*)alloc((size_t)kHD * kN * 2);
  unsigned short* og   = (unsigned short*)alloc((size_t)kN * kHD * 2);
  float*          gate = (float*)alloc((size_t)kN * kHD * 4);
  float*          pb   = (float*)alloc((size_t)kH * kN * kN * 4);   // 64 MB (L2-resident)

  const int wElems = kCQ * kHD;
  const int wBlocks = (wElems + 255) / 256;
  k_wconv<<<wBlocks, 256, 0, stream>>>(w_q, wqt, kCQ, kHD);
  k_wconv<<<wBlocks, 256, 0, stream>>>(w_k, wkt, kCQ, kHD);
  k_wconv<<<wBlocks, 256, 0, stream>>>(w_v, wvt, kCQ, kHD);
  k_wconv<<<wBlocks, 256, 0, stream>>>(w_g, wgt, kCQ, kHD);
  k_wconv<<<wBlocks, 256, 0, stream>>>(w_o, wot, kHD, kCQ);

  k_ln_a<<<kN, 256, 0, stream>>>(a, g_a, b_a, an);

  k_proj<<<dim3(kHD / 16, kN / 16, 4), 32, 0, stream>>>(an, wqt, wkt, wvt, wgt,
                                                        b_g, qb, kb, vt, gate);

  k_pairbias<<<(kN * kN) / 8, 256, 0, stream>>>(z, g_z, b_z, w_z, pb);

  k_attn<<<dim3(kN / 16, kH), 256, 0, stream>>>(qb, kb, vt, pb, mask, gate, og);

  k_outgemm<<<dim3(kCQ / 16, kN / 16), 32, 0, stream>>>(og, wot, b_o, out);
}
